// MHSAttention_41987600286090
// MI455X (gfx1250) — matrix operand order repaired and checked by reading, not compiled, not run on hardware
//
#include <hip/hip_runtime.h>

// ---------------------------------------------------------------------------
// MHSA forward for MI455X (gfx1250, wave32, WMMA).
//  qkv = x@Wqkv^T+b ; softmax over QUERY axis (per-column of A=QK^T/8) ; P@V ;
//  out = MSA@Wo^T+bo.  GEMMs via v_wmma_f32_16x16x32_bf16; tile staging via
//  global_load_async_to_lds_b128 (ASYNCcnt) and TDM tensor_load_to_lds
//  (TENSORcnt), ping-pong double-buffered so the DMA overlaps the WMMAs.
// ---------------------------------------------------------------------------

typedef __bf16 bf16;
typedef __attribute__((ext_vector_type(16))) __bf16 bf16x16;
typedef __attribute__((ext_vector_type(8)))  __bf16 bf16x8;
typedef __attribute__((ext_vector_type(8)))  float  floatx8;
typedef __attribute__((ext_vector_type(4)))  float  floatx4;
typedef __attribute__((ext_vector_type(4)))  int    intx4;
typedef __attribute__((ext_vector_type(8)))  int    intx8;
typedef __attribute__((ext_vector_type(4)))  unsigned int uintx4;

#define DIMM   768
#define HEADS  12
#define DH     64
#define BATCH  16
#define SEQ    1024
#define M_TOK  (BATCH*SEQ)      /* 16384 */
#define N_QKV  (3*DIMM)         /* 2304  */
#define SCALEA 0.125f           /* 1/sqrt(64) */

// ---- CDNA5 async / TDM feature detection ----------------------------------
#if __has_builtin(__builtin_amdgcn_global_load_async_to_lds_b128) && \
    __has_builtin(__builtin_amdgcn_s_wait_asynccnt)
#define HAVE_ASYNC_LDS 1
#else
#define HAVE_ASYNC_LDS 0
#endif

#if __has_builtin(__builtin_amdgcn_tensor_load_to_lds) && \
    __has_builtin(__builtin_amdgcn_s_wait_tensorcnt)
#define HAVE_TDM 1
#else
#define HAVE_TDM 0
#endif

// 16B global->LDS tile copy, async when available (per-lane addresses).
// Builtin takes pointers to int __vector(4): global src AS1, LDS dst AS3.
__device__ inline void copy16(void* lds, const void* g) {
#if HAVE_ASYNC_LDS
  auto gp = (__attribute__((address_space(1))) intx4*)(unsigned long long)g;
  auto lp = (__attribute__((address_space(3))) intx4*)
            (unsigned int)(unsigned long long)lds;
  __builtin_amdgcn_global_load_async_to_lds_b128(gp, lp, 0, 0);
#else
  *(bf16x8*)lds = *(const bf16x8*)g;
#endif
}

__device__ inline void async_wait() {
#if HAVE_ASYNC_LDS
  __builtin_amdgcn_s_wait_asynccnt(0);
#endif
}

#if HAVE_TDM
// TDM 2D tile load (CDNA5 ISA). Tensor dims == tile
// dims (tiles are interior, no OOB), data_size = 2 bytes (bf16).
// tile_d0 = contiguous elements per row, tile_d1 = rows, stride0 in elems.
__device__ inline void tdm_load_2d(unsigned lds_off, const void* gptr,
                                   unsigned tile_d0, unsigned tile_d1,
                                   unsigned stride0) {
  unsigned long long ga = (unsigned long long)gptr;
  uintx4 g0;
  g0[0] = 1u;                                   // count=1, user descriptor
  g0[1] = lds_off;                              // lds_addr (bytes)
  g0[2] = (unsigned)(ga & 0xffffffffu);         // global_addr[31:0]
  g0[3] = (unsigned)((ga >> 32) & 0x01ffffffu)  // global_addr[56:32]
          | (2u << 30);                         // type=2 ("image")
  intx8 g1;
  g1[0] = (int)(1u << 16);                      // data_size=1 -> 2 bytes
  g1[1] = (int)((tile_d0 & 0xffffu) << 16);     // tensor_dim0[15:0]
  g1[2] = (int)(((tile_d0 >> 16) & 0xffffu)     // tensor_dim0[31:16]
          | ((tile_d1 & 0xffffu) << 16));       // tensor_dim1[15:0]
  g1[3] = (int)(((tile_d1 >> 16) & 0xffffu)     // tensor_dim1[31:16]
          | ((tile_d0 & 0xffffu) << 16));       // tile_dim0
  g1[4] = (int)(tile_d1 & 0xffffu);             // tile_dim1 | tile_dim2=0
  g1[5] = (int)stride0;                         // tensor_dim0_stride[31:0]
  g1[6] = 0;                                    // stride0[47:32]|stride1 lo
  g1[7] = 0;
  intx4 z4 = (intx4)0;
#if __clang_major__ >= 23
  intx8 z8 = (intx8)0;
  __builtin_amdgcn_tensor_load_to_lds(g0, g1, z4, z4, z8, 0);
#else
  __builtin_amdgcn_tensor_load_to_lds(g0, g1, z4, z4, 0);
#endif
}
#endif

// ---- WMMA fragment helpers (contiguous k order k = 16*(L/16)+i, wave32) -----

// A-matrix 16x32 bf16: lane L holds row m=L%16; elems i = K (L/16)*16+i.
// Two contiguous 16B loads.
__device__ inline bf16x16 load_a_frag(const bf16* base, int ld) {
  int lane = threadIdx.x & 31;
  int m  = lane & 15;
  int kb = (lane >> 4) << 4;
  const bf16* p = base + m * ld + kb;
  bf16x8 lo = *(const bf16x8*)(p);
  bf16x8 hi = *(const bf16x8*)(p + 8);
  bf16x16 r;
#pragma unroll
  for (int i = 0; i < 8; ++i) { r[i] = lo[i]; r[8 + i] = hi[i]; }
  return r;
}

// B-matrix 32x16 bf16 (B = rowmajor[n][k]): lane L holds col n=L%16;
// elems i = K (L/16)*16+i.  One contiguous 32B load.
__device__ inline bf16x16 load_b_frag(const bf16* base, int ld) {
  int lane = threadIdx.x & 31;
  int n  = lane & 15;
  int kb = (lane >> 4) << 4;
  const bf16* p = base + n * ld + kb;
  bf16x8 lo = *(const bf16x8*)(p);
  bf16x8 hi = *(const bf16x8*)(p + 8);
  bf16x16 r;
#pragma unroll
  for (int i = 0; i < 8; ++i) { r[i] = lo[i]; r[8 + i] = hi[i]; }
  return r;
}

__device__ inline floatx8 wmma_bf16(bf16x16 a, bf16x16 b, floatx8 c) {
  return __builtin_amdgcn_wmma_f32_16x16x32_bf16(
      /*neg_a=*/false, a, /*neg_b=*/false, b,
      /*c_mod=*/(short)0, c, /*reuse_a=*/false, /*reuse_b=*/false);
}

// ---------------------------------------------------------------------------
// 0) f32 -> bf16 conversion for x, Wqkv, Wo
// ---------------------------------------------------------------------------
__global__ __launch_bounds__(256) void convert_bf16_kernel(
    const float* __restrict__ x, const float* __restrict__ wqkv,
    const float* __restrict__ wo,
    bf16* __restrict__ xb, bf16* __restrict__ wqkvb, bf16* __restrict__ wob) {
  const int n1 = M_TOK * DIMM, n2 = N_QKV * DIMM, n3 = DIMM * DIMM;
  int idx = blockIdx.x * 256 + threadIdx.x;
  if (idx < n1)                xb[idx]             = (bf16)x[idx];
  else if (idx < n1 + n2)      wqkvb[idx - n1]     = (bf16)wqkv[idx - n1];
  else if (idx < n1 + n2 + n3) wob[idx - n1 - n2]  = (bf16)wo[idx - n1 - n2];
}

// ---------------------------------------------------------------------------
// 1) QKV GEMM: C[m,n] = sum_k xb[m,k]*Wqkv[n,k] + bqkv[n]
//    128x128 tile, Kstep=32, 8 waves (4Mx2N), wave tile 32x64.
//    Double-buffered async-LDS pipeline: fill tile t+1 while computing t.
//    Epilogue scatters into head-major Q [bh,p,d], K [bh,p,d], V^T [bh,d,p].
// ---------------------------------------------------------------------------
__global__ __launch_bounds__(256) void gemm_qkv_kernel(
    const bf16* __restrict__ xb, const bf16* __restrict__ wb,
    const float* __restrict__ bias,
    bf16* __restrict__ qout, bf16* __restrict__ kout, bf16* __restrict__ vtout) {
  __shared__ bf16 sx[2][128 * 32];
  __shared__ bf16 sw[2][128 * 32];
  int tid = threadIdx.x, lane = tid & 31, wave = tid >> 5;
  int m0 = blockIdx.x * 128, n0 = blockIdx.y * 128;
  int wm = wave >> 1, wn = wave & 1;

  floatx8 acc[2][4];
#pragma unroll
  for (int mi = 0; mi < 2; ++mi)
#pragma unroll
    for (int ni = 0; ni < 4; ++ni) acc[mi][ni] = (floatx8)0.0f;

  int r = tid >> 1, c = (tid & 1) << 4;
  auto issue_tile = [&](int buf, int k0) {
    copy16(&sx[buf][r * 32 + c],     &xb[(size_t)(m0 + r) * DIMM + k0 + c]);
    copy16(&sx[buf][r * 32 + c + 8], &xb[(size_t)(m0 + r) * DIMM + k0 + c + 8]);
    copy16(&sw[buf][r * 32 + c],     &wb[(size_t)(n0 + r) * DIMM + k0 + c]);
    copy16(&sw[buf][r * 32 + c + 8], &wb[(size_t)(n0 + r) * DIMM + k0 + c + 8]);
  };

  issue_tile(0, 0);
  int cur = 0;
  for (int k0 = 0; k0 < DIMM; k0 += 32) {
    async_wait();        // current buffer's fill complete (this wave)
    __syncthreads();     // all waves' fills complete / readers of nxt done
    if (k0 + 32 < DIMM) issue_tile(cur ^ 1, k0 + 32);  // overlap with WMMAs
    const bf16* cx = sx[cur];
    const bf16* cw = sw[cur];
    bf16x16 af[2], bfr[4];
#pragma unroll
    for (int mi = 0; mi < 2; ++mi) af[mi]  = load_a_frag(cx + (wm * 32 + mi * 16) * 32, 32);
#pragma unroll
    for (int ni = 0; ni < 4; ++ni) bfr[ni] = load_b_frag(cw + (wn * 64 + ni * 16) * 32, 32);
#pragma unroll
    for (int mi = 0; mi < 2; ++mi)
#pragma unroll
      for (int ni = 0; ni < 4; ++ni) acc[mi][ni] = wmma_bf16(af[mi], bfr[ni], acc[mi][ni]);
    cur ^= 1;
  }

  // C-frag layout: lane L -> col n = L%16 ; elem e -> row m = e + 8*(L/16)
#pragma unroll
  for (int ni = 0; ni < 4; ++ni) {
    int n = n0 + wn * 64 + ni * 16 + (lane & 15);
    float bv = bias[n];
#pragma unroll
    for (int mi = 0; mi < 2; ++mi) {
#pragma unroll
      for (int e = 0; e < 8; ++e) {
        int m  = m0 + wm * 32 + mi * 16 + e + ((lane >> 4) << 3);
        float v = acc[mi][ni][e] + bv;
        int bb = m >> 10, p = m & 1023;
        if (n < DIMM) {
          int h = n >> 6, d = n & 63;
          qout[(((size_t)bb * HEADS + h) * SEQ + p) * DH + d] = (bf16)v;
        } else if (n < 2 * DIMM) {
          int nn = n - DIMM, h = nn >> 6, d = nn & 63;
          kout[(((size_t)bb * HEADS + h) * SEQ + p) * DH + d] = (bf16)v;
        } else {
          int nn = n - 2 * DIMM, h = nn >> 6, d = nn & 63;
          vtout[(((size_t)bb * HEADS + h) * DH + d) * SEQ + p] = (bf16)v;
        }
      }
    }
  }
}

// ---------------------------------------------------------------------------
// 2) Column softmax stats: for each (b,h,k): M[k]=max_q s*A[q,k],
//    Z[k]=sum_q exp(s*A[q,k]-M[k]).  Online over q-tiles of 16.
//    S = Q K^T orientation so the reduction is over C-frag rows (cheap:
//    8 in-lane elems + one shfl_xor(16)).
// ---------------------------------------------------------------------------
__global__ __launch_bounds__(256) void attn_stats_kernel(
    const bf16* __restrict__ qb, const bf16* __restrict__ kbuf,
    float* __restrict__ Mst, float* __restrict__ Zst) {
  int lane = threadIdx.x & 31, wave = threadIdx.x >> 5;
  int bh = blockIdx.z * HEADS + blockIdx.y;
  const bf16* Q = qb   + (size_t)bh * SEQ * DH;
  const bf16* K = kbuf + (size_t)bh * SEQ * DH;
  int c0 = blockIdx.x * 128 + wave * 16;   // this wave's 16 key-columns

  bf16x16 kf0 = load_b_frag(K + (size_t)c0 * DH, DH);       // B = K^T, d 0..31
  bf16x16 kf1 = load_b_frag(K + (size_t)c0 * DH + 32, DH);  // d 32..63

  float mr = -__builtin_inff();
  float lr = 0.0f;
  for (int q0 = 0; q0 < SEQ; q0 += 16) {
    if (q0 + 16 < SEQ)  // prefetch next Q tile -> global_prefetch_b8
      __builtin_prefetch(Q + (size_t)(q0 + 16) * DH + (lane << 5), 0, 3);
    bf16x16 a0 = load_a_frag(Q + (size_t)q0 * DH, DH);
    bf16x16 a1 = load_a_frag(Q + (size_t)q0 * DH + 32, DH);
    floatx8 s = (floatx8)0.0f;
    s = wmma_bf16(a0, kf0, s);
    s = wmma_bf16(a1, kf1, s);
    float tm = -__builtin_inff();
#pragma unroll
    for (int e = 0; e < 8; ++e) tm = fmaxf(tm, SCALEA * s[e]);
    float nm = fmaxf(mr, tm);
    float add = 0.0f;
#pragma unroll
    for (int e = 0; e < 8; ++e) add += __expf(SCALEA * s[e] - nm);
    lr = lr * __expf(mr - nm) + add;
    mr = nm;
  }
  // merge the two row-halves (lane L <-> L+16 hold the same column)
  float om = __shfl_xor(mr, 16, 32);
  float ol = __shfl_xor(lr, 16, 32);
  float gm = fmaxf(mr, om);
  float gl = lr * __expf(mr - gm) + ol * __expf(om - gm);
  if (lane < 16) {
    Mst[(size_t)bh * SEQ + c0 + lane] = gm;
    Zst[(size_t)bh * SEQ + c0 + lane] = gl;
  }
}

// ---------------------------------------------------------------------------
// 3) Fused normalize + PV:  out^T[d,q] = sum_k V^T[d,k] * P^T[k,q]
//    S^T = K Q^T is computed so its C-frag (lane = q column) converts to the
//    PV B-frag with a single shfl_xor(16) half-swap.  K / V^T tiles staged
//    into LDS by the Tensor Data Mover (wave 0 issues, TENSORcnt tracked),
//    ping-pong double-buffered so the TDM overlaps the WMMAs.
// ---------------------------------------------------------------------------
__global__ __launch_bounds__(256) void attn_pv_kernel(
    const bf16* __restrict__ qb, const bf16* __restrict__ kbuf,
    const bf16* __restrict__ vtb, const float* __restrict__ Mst,
    const float* __restrict__ Zst, bf16* __restrict__ msab) {
  __shared__ bf16 skt[2][32 * 64];   // K rows   [32k x 64d]
  __shared__ bf16 svt[2][64 * 32];   // V^T tile [64d x 32k]
  int tid = threadIdx.x, lane = tid & 31, wave = tid >> 5;
  int b = blockIdx.z, h = blockIdx.y;
  int bh = b * HEADS + h;
  const bf16* Q  = qb   + (size_t)bh * SEQ * DH;
  const bf16* K  = kbuf + (size_t)bh * SEQ * DH;
  const bf16* Vt = vtb  + (size_t)bh * DH * SEQ;
  const float* Mc = Mst + (size_t)bh * SEQ;
  const float* Zc = Zst + (size_t)bh * SEQ;
  int q0  = blockIdx.x * 128 + wave * 16;   // this wave's 16 queries
  int sel = lane >> 4;

  // Q^T B-frags (loop invariant): B[dd,n] = Q[q0+n, dd]
  bf16x16 qf0 = load_b_frag(Q + (size_t)q0 * DH, DH);
  bf16x16 qf1 = load_b_frag(Q + (size_t)q0 * DH + 32, DH);

  floatx8 o[4];
#pragma unroll
  for (int dt = 0; dt < 4; ++dt) o[dt] = (floatx8)0.0f;

#if HAVE_TDM
  auto issue_tile = [&](int buf, int k0) {
    tdm_load_2d((unsigned)(size_t)(void*)&skt[buf][0], K + (size_t)k0 * DH,
                /*tile_d0=*/64, /*tile_d1=*/32, /*stride0=*/DH);
    tdm_load_2d((unsigned)(size_t)(void*)&svt[buf][0], Vt + k0,
                /*tile_d0=*/32, /*tile_d1=*/64, /*stride0=*/SEQ);
  };
  if (wave == 0) issue_tile(0, 0);
#else
  int kr = tid >> 3, kc = (tid & 7) << 3;   // K-tile copy coords
  int vr = tid >> 2, vc = (tid & 3) << 3;   // Vt-tile copy coords
  auto issue_tile = [&](int buf, int k0) {
    copy16(&skt[buf][kr * 64 + kc], &K[(size_t)(k0 + kr) * DH + kc]);
    copy16(&svt[buf][vr * 32 + vc], &Vt[(size_t)vr * SEQ + k0 + vc]);
  };
  issue_tile(0, 0);
#endif

  int cur = 0;
  for (int k0 = 0; k0 < SEQ; k0 += 32) {
#if HAVE_TDM
    if (wave == 0) __builtin_amdgcn_s_wait_tensorcnt(0);
    __syncthreads();
    if (wave == 0 && k0 + 32 < SEQ) issue_tile(cur ^ 1, k0 + 32);
#else
    async_wait();
    __syncthreads();
    if (k0 + 32 < SEQ) issue_tile(cur ^ 1, k0 + 32);
#endif
    const bf16* ck = skt[cur];
    const bf16* cv = svt[cur];

    // S^T frags [16k x 16q] for the two k sub-tiles
    floatx8 st[2];
#pragma unroll
    for (int kt = 0; kt < 2; ++kt) {
      floatx8 s = (floatx8)0.0f;
      s = wmma_bf16(load_a_frag(ck + (kt * 16) * 64, 64),      qf0, s);
      s = wmma_bf16(load_a_frag(ck + (kt * 16) * 64 + 32, 64), qf1, s);
      st[kt] = s;
    }

    // p = exp(s*A - M[k]) / Z[k]; elem e -> k = k0 + kt*16 + e + 8*sel
    bf16x8 own[2];
#pragma unroll
    for (int kt = 0; kt < 2; ++kt) {
      int kkb = k0 + kt * 16 + (sel << 3);
      floatx4 mlo = *(const floatx4*)&Mc[kkb];
      floatx4 mhi = *(const floatx4*)&Mc[kkb + 4];
      floatx4 zlo = *(const floatx4*)&Zc[kkb];
      floatx4 zhi = *(const floatx4*)&Zc[kkb + 4];
#pragma unroll
      for (int e = 0; e < 8; ++e) {
        float mv = (e < 4) ? mlo[e] : mhi[e - 4];
        float zv = (e < 4) ? zlo[e] : zhi[e - 4];
        float pv = __expf(SCALEA * st[kt][e] - mv) * (1.0f / zv);
        own[kt][e] = (bf16)pv;
      }
    }
    // C-frag -> B-frag: low lanes take [kt0.own | kt0.partner],
    //                   high lanes take [kt1.partner | kt1.own]
    intx4 owni0 = __builtin_bit_cast(intx4, own[0]);
    intx4 owni1 = __builtin_bit_cast(intx4, own[1]);
    intx4 othi0, othi1;
#pragma unroll
    for (int j = 0; j < 4; ++j) {
      othi0[j] = __shfl_xor(owni0[j], 16, 32);
      othi1[j] = __shfl_xor(owni1[j], 16, 32);
    }
    intx8 comb;
#pragma unroll
    for (int j = 0; j < 4; ++j) {
      comb[j]     = sel ? othi1[j] : owni0[j];
      comb[4 + j] = sel ? owni1[j] : othi0[j];
    }
    bf16x16 pf = __builtin_bit_cast(bf16x16, comb);

    // out^T[d,q] += Vt[16d x 32k] * P^T[32k x 16q]
#pragma unroll
    for (int dt = 0; dt < 4; ++dt)
      o[dt] = wmma_bf16(load_a_frag(cv + (dt * 16) * 32, 32), pf, o[dt]);
    cur ^= 1;
  }

  // epilogue: lane -> q col; elem e -> d row (consecutive) => 16B stores
  int q = q0 + (lane & 15);
#pragma unroll
  for (int dt = 0; dt < 4; ++dt) {
    bf16x8 ov;
#pragma unroll
    for (int e = 0; e < 8; ++e) ov[e] = (bf16)o[dt][e];
    int d = dt * 16 + (sel << 3);
    *(bf16x8*)&msab[((size_t)(b * SEQ + q)) * DIMM + h * DH + d] = ov;
  }
}

// ---------------------------------------------------------------------------
// 4) Output projection: out[m,n] = sum_k MSA[m,k]*Wo[n,k] + bo[n]  (f32 out)
//    Same double-buffered async-LDS pipeline as the QKV GEMM.
// ---------------------------------------------------------------------------
__global__ __launch_bounds__(256) void gemm_out_kernel(
    const bf16* __restrict__ ab, const bf16* __restrict__ wb,
    const float* __restrict__ bias, float* __restrict__ out) {
  __shared__ bf16 sx[2][128 * 32];
  __shared__ bf16 sw[2][128 * 32];
  int tid = threadIdx.x, lane = tid & 31, wave = tid >> 5;
  int m0 = blockIdx.x * 128, n0 = blockIdx.y * 128;
  int wm = wave >> 1, wn = wave & 1;

  floatx8 acc[2][4];
#pragma unroll
  for (int mi = 0; mi < 2; ++mi)
#pragma unroll
    for (int ni = 0; ni < 4; ++ni) acc[mi][ni] = (floatx8)0.0f;

  int r = tid >> 1, c = (tid & 1) << 4;
  auto issue_tile = [&](int buf, int k0) {
    copy16(&sx[buf][r * 32 + c],     &ab[(size_t)(m0 + r) * DIMM + k0 + c]);
    copy16(&sx[buf][r * 32 + c + 8], &ab[(size_t)(m0 + r) * DIMM + k0 + c + 8]);
    copy16(&sw[buf][r * 32 + c],     &wb[(size_t)(n0 + r) * DIMM + k0 + c]);
    copy16(&sw[buf][r * 32 + c + 8], &wb[(size_t)(n0 + r) * DIMM + k0 + c + 8]);
  };

  issue_tile(0, 0);
  int cur = 0;
  for (int k0 = 0; k0 < DIMM; k0 += 32) {
    async_wait();
    __syncthreads();
    if (k0 + 32 < DIMM) issue_tile(cur ^ 1, k0 + 32);
    const bf16* cx = sx[cur];
    const bf16* cw = sw[cur];
    bf16x16 af[2], bfr[4];
#pragma unroll
    for (int mi = 0; mi < 2; ++mi) af[mi]  = load_a_frag(cx + (wm * 32 + mi * 16) * 32, 32);
#pragma unroll
    for (int ni = 0; ni < 4; ++ni) bfr[ni] = load_b_frag(cw + (wn * 64 + ni * 16) * 32, 32);
#pragma unroll
    for (int mi = 0; mi < 2; ++mi)
#pragma unroll
      for (int ni = 0; ni < 4; ++ni) acc[mi][ni] = wmma_bf16(af[mi], bfr[ni], acc[mi][ni]);
    cur ^= 1;
  }

#pragma unroll
  for (int ni = 0; ni < 4; ++ni) {
    int n = n0 + wn * 64 + ni * 16 + (lane & 15);
    float bv = bias[n];
#pragma unroll
    for (int mi = 0; mi < 2; ++mi)
#pragma unroll
      for (int e = 0; e < 8; ++e) {
        int m = m0 + wm * 32 + mi * 16 + e + ((lane >> 4) << 3);
        out[(size_t)m * DIMM + n] = acc[mi][ni][e] + bv;
      }
  }
}

// ---------------------------------------------------------------------------
extern "C" void kernel_launch(void* const* d_in, const int* in_sizes, int n_in,
                              void* d_out, int out_size, void* d_ws, size_t ws_size,
                              hipStream_t stream) {
  (void)in_sizes; (void)n_in; (void)out_size; (void)ws_size;
  const float* x    = (const float*)d_in[0];
  const float* Wqkv = (const float*)d_in[1];
  const float* bqkv = (const float*)d_in[2];
  const float* Wo   = (const float*)d_in[3];
  const float* bo   = (const float*)d_in[4];
  float* out = (float*)d_out;

  // workspace carve (~133 MB total), 256B aligned regions
  char* ws = (char*)d_ws;
  size_t off = 0;
  auto carve = [&](size_t bytes) -> char* {
    char* p = ws + off;
    off += (bytes + 255) & ~(size_t)255;
    return p;
  };
  bf16* xb    = (bf16*)carve((size_t)M_TOK * DIMM * 2);
  bf16* wqkvb = (bf16*)carve((size_t)N_QKV * DIMM * 2);
  bf16* wob   = (bf16*)carve((size_t)DIMM * DIMM * 2);
  bf16* qb    = (bf16*)carve((size_t)BATCH * HEADS * SEQ * DH * 2);
  bf16* kbuf  = (bf16*)carve((size_t)BATCH * HEADS * SEQ * DH * 2);
  bf16* vtb   = (bf16*)carve((size_t)BATCH * HEADS * DH * SEQ * 2);
  float* Mst  = (float*)carve((size_t)BATCH * HEADS * SEQ * 4);
  float* Zst  = (float*)carve((size_t)BATCH * HEADS * SEQ * 4);
  bf16* msab  = (bf16*)carve((size_t)M_TOK * DIMM * 2);

  const int total_cvt = M_TOK * DIMM + N_QKV * DIMM + DIMM * DIMM;
  convert_bf16_kernel<<<(total_cvt + 255) / 256, 256, 0, stream>>>(
      x, Wqkv, Wo, xb, wqkvb, wob);

  gemm_qkv_kernel<<<dim3(M_TOK / 128, N_QKV / 128), 256, 0, stream>>>(
      xb, wqkvb, bqkv, qb, kbuf, vtb);

  attn_stats_kernel<<<dim3(SEQ / 128, HEADS, BATCH), 256, 0, stream>>>(
      qb, kbuf, Mst, Zst);

  attn_pv_kernel<<<dim3(SEQ / 128, HEADS, BATCH), 256, 0, stream>>>(
      qb, kbuf, vtb, Mst, Zst, msab);

  gemm_out_kernel<<<dim3(M_TOK / 128, DIMM / 128), 256, 0, stream>>>(
      msab, wob, bo, out);
}
